// Reg2DDecode3D_32366873543225
// MI455X (gfx1250) — matrix operand order repaired and checked by reading, not compiled, not run on hardware
//
#include <hip/hip_runtime.h>
#include <hip/hip_bf16.h>

typedef _Float16 v16h __attribute__((ext_vector_type(16)));
typedef _Float16 v8h  __attribute__((ext_vector_type(8)));
typedef float    v8f  __attribute__((ext_vector_type(8)));

// ---------------------------------------------------------------------------
// f32 -> f16 weight conversion
// ---------------------------------------------------------------------------
__global__ void cvt_w_kernel(const float* __restrict__ src, _Float16* __restrict__ dst, int n) {
    int i = blockIdx.x * blockDim.x + threadIdx.x;
    if (i < n) dst[i] = (_Float16)src[i];
}

// ---------------------------------------------------------------------------
// Bilinear grid-sample of 21 uv points per batch over 512 channels.
// h0: (B=64, 21, 512) f32
// ---------------------------------------------------------------------------
__global__ void sample_kernel(const float* __restrict__ uv, const float* __restrict__ x,
                              float* __restrict__ h0) {
    int n = blockIdx.x;   // point 0..20
    int b = blockIdx.y;   // batch 0..63
    float u  = uv[(b * 21 + n) * 2 + 0];
    float vv = uv[(b * 21 + n) * 2 + 1];
    float g0 = fminf(fmaxf((u  - 0.5f) * 2.0f, -1.0f), 1.0f);
    float g1 = fminf(fmaxf((vv - 0.5f) * 2.0f, -1.0f), 1.0f);
    float gx = (g0 + 1.0f) * 0.5f * 63.0f;
    float gy = (g1 + 1.0f) * 0.5f * 63.0f;
    float x0f = floorf(gx), y0f = floorf(gy);
    float wx1 = gx - x0f, wy1 = gy - y0f;
    float wx0 = 1.0f - wx1, wy0 = 1.0f - wy1;
    int x0 = min(max((int)x0f, 0), 63), x1 = min(max((int)x0f + 1, 0), 63);
    int y0 = min(max((int)y0f, 0), 63), y1 = min(max((int)y0f + 1, 0), 63);
    float wa = wx0 * wy0, wb = wx0 * wy1, wc = wx1 * wy0, wd = wx1 * wy1;
    int ia = y0 * 64 + x0, ib = y1 * 64 + x0, ic = y0 * 64 + x1, id = y1 * 64 + x1;
    for (int c = threadIdx.x; c < 512; c += blockDim.x) {
        const float* F = x + ((size_t)(b * 512 + c)) * 4096;
        h0[((size_t)b * 21 + n) * 512 + c] = wa * F[ia] + wb * F[ib] + wc * F[ic] + wd * F[id];
    }
}

// ---------------------------------------------------------------------------
// h1[b][v][c] = sum_n up[v][n] * h0[b][n][c]     (49 x 21 dense)  f32 out
// ---------------------------------------------------------------------------
__global__ void upsample_kernel(const float* __restrict__ up, const float* __restrict__ h0,
                                float* __restrict__ h1) {
    int v = blockIdx.x;   // 0..48
    int b = blockIdx.y;   // 0..63
    for (int c = threadIdx.x; c < 512; c += blockDim.x) {
        float s = 0.0f;
        for (int n = 0; n < 21; ++n)
            s += up[v * 21 + n] * h0[((size_t)b * 21 + n) * 512 + c];
        h1[((size_t)b * 49 + v) * 512 + c] = s;
    }
}

// ---------------------------------------------------------------------------
// Sparse pool (3 nnz/row): out16[b][v][c] = f16( sum_k val[3v+k]*in[b][col[3v+k]][c] )
// ---------------------------------------------------------------------------
__global__ void pool_kernel(const float* __restrict__ hin, const int* __restrict__ idx,
                            const float* __restrict__ val, _Float16* __restrict__ hout,
                            int Vin, int Vout, int C) {
    int v = blockIdx.x;
    int b = blockIdx.y;
    int base = 3 * v;
    int c0 = idx[(base + 0) * 2 + 1];
    int c1 = idx[(base + 1) * 2 + 1];
    int c2 = idx[(base + 2) * 2 + 1];
    float w0 = val[base], w1 = val[base + 1], w2 = val[base + 2];
    const float* r0 = hin + ((size_t)b * Vin + c0) * C;
    const float* r1 = hin + ((size_t)b * Vin + c1) * C;
    const float* r2 = hin + ((size_t)b * Vin + c2) * C;
    _Float16* o = hout + ((size_t)b * Vout + v) * C;
    for (int c = threadIdx.x; c < C; c += blockDim.x)
        o[c] = (_Float16)(w0 * r0[c] + w1 * r1[c] + w2 * r2[c]);
}

// ---------------------------------------------------------------------------
// Spiral-gather GEMM with WMMA f32<-f16:
//   out[b][v][n] = relu( bias[n] + sum_{l,c} A16[b][sp[v][l]][c] * Wt[n][l*C+c] )
// One wave computes a 32(M) x 64(N) tile: two A fragments share each of the
// four B fragments -> 12 b128 loads per 8 WMMAs. K consumed in chunks of 32.
// M = 64*Vlvl (mult of 32), N mult of 64, C mult of 32 (>=64) so a 32-chunk
// never crosses a spiral-neighbor boundary.
// A/B fragments use contiguous K (lane half hi: 16*hi..16*hi+15); D uses the ISA layout (wave32).
// ---------------------------------------------------------------------------
__global__ __launch_bounds__(256)
void gemm_spiral_kernel(const _Float16* __restrict__ A, const int* __restrict__ sp,
                        const _Float16* __restrict__ Wt, const float* __restrict__ bias,
                        float* __restrict__ out, int Vlvl, int C, int N,
                        int mtiles32, int ntiles64) {
    int wave = __builtin_amdgcn_readfirstlane((int)(threadIdx.x >> 5));
    int tile = __builtin_amdgcn_readfirstlane((int)blockIdx.x) * 8 + wave;
    int total = mtiles32 * ntiles64;
    if (tile >= total) return;            // scalar branch: EXEC stays all-1s

    int tm = tile / ntiles64;
    int tn = tile - tm * ntiles64;
    int lane  = (int)(threadIdx.x & 31);
    int lhalf = lane & 15;
    int hi    = lane >> 4;                // 0 for lanes 0-15, 1 for lanes 16-31
    int K = 9 * C;

    // A-fragment source rows for this lane (16-bit A 16x32 layout):
    // fragment 0 covers output rows tm*32 .. tm*32+15, fragment 1 covers +16.
    int m0r = tm * 32 + lhalf;
    int m1r = m0r + 16;
    int b0r = m0r / Vlvl, v0r = m0r - b0r * Vlvl;
    int b1r = m1r / Vlvl, v1r = m1r - b1r * Vlvl;
    const int* sprow0 = sp + v0r * 9;
    const int* sprow1 = sp + v1r * 9;
    const _Float16* abase0 = A + (size_t)b0r * Vlvl * C;
    const _Float16* abase1 = A + (size_t)b1r * Vlvl * C;
    int aoff = hi ? 16 : 0;               // K offsets: aoff+0..15 (contiguous)

    // B-fragment: column n, 16 consecutive K starting at kbase
    int n0    = tn * 64 + lhalf;
    int kbase = hi ? 16 : 0;

    v8f a0c0, a0c1, a0c2, a0c3;           // fragment-0 accumulators (rows 0..15)
    v8f a1c0, a1c1, a1c2, a1c3;           // fragment-1 accumulators (rows 16..31)
    float bz0 = bias[n0], bz1 = bias[n0 + 16], bz2 = bias[n0 + 32], bz3 = bias[n0 + 48];
    #pragma unroll
    for (int i = 0; i < 8; ++i) {
        a0c0[i] = bz0; a0c1[i] = bz1; a0c2[i] = bz2; a0c3[i] = bz3;
        a1c0[i] = bz0; a1c1[i] = bz1; a1c2[i] = bz2; a1c3[i] = bz3;
    }

    for (int l = 0; l < 9; ++l) {
        const _Float16* arow0 = abase0 + (size_t)sprow0[l] * C;
        const _Float16* arow1 = abase1 + (size_t)sprow1[l] * C;
        if (l + 1 < 9) {                  // cover gather latency of next neighbor
            __builtin_prefetch(abase0 + (size_t)sprow0[l + 1] * C, 0, 3);
            __builtin_prefetch(abase1 + (size_t)sprow1[l + 1] * C, 0, 3);
        }
        for (int c0 = 0; c0 < C; c0 += 32) {
            v8h alo = *(const v8h*)(arow0 + c0 + aoff);
            v8h ahi = *(const v8h*)(arow0 + c0 + aoff + 8);
            v16h af0 = __builtin_shufflevector(alo, ahi,
                0, 1, 2, 3, 4, 5, 6, 7, 8, 9, 10, 11, 12, 13, 14, 15);
            alo = *(const v8h*)(arow1 + c0 + aoff);
            ahi = *(const v8h*)(arow1 + c0 + aoff + 8);
            v16h af1 = __builtin_shufflevector(alo, ahi,
                0, 1, 2, 3, 4, 5, 6, 7, 8, 9, 10, 11, 12, 13, 14, 15);

            int k = l * C + c0 + kbase;
            const _Float16* w0 = Wt + (size_t)n0 * K + k;

            v8h t0 = *(const v8h*)w0;
            v8h t1 = *(const v8h*)(w0 + 8);
            v16h bf = __builtin_shufflevector(t0, t1,
                0, 1, 2, 3, 4, 5, 6, 7, 8, 9, 10, 11, 12, 13, 14, 15);
            a0c0 = __builtin_amdgcn_wmma_f32_16x16x32_f16(false, af0, false, bf,
                                                          (short)0, a0c0, false, false);
            a1c0 = __builtin_amdgcn_wmma_f32_16x16x32_f16(false, af1, false, bf,
                                                          (short)0, a1c0, false, false);

            t0 = *(const v8h*)(w0 + (size_t)16 * K);
            t1 = *(const v8h*)(w0 + (size_t)16 * K + 8);
            bf = __builtin_shufflevector(t0, t1,
                0, 1, 2, 3, 4, 5, 6, 7, 8, 9, 10, 11, 12, 13, 14, 15);
            a0c1 = __builtin_amdgcn_wmma_f32_16x16x32_f16(false, af0, false, bf,
                                                          (short)0, a0c1, false, false);
            a1c1 = __builtin_amdgcn_wmma_f32_16x16x32_f16(false, af1, false, bf,
                                                          (short)0, a1c1, false, false);

            t0 = *(const v8h*)(w0 + (size_t)32 * K);
            t1 = *(const v8h*)(w0 + (size_t)32 * K + 8);
            bf = __builtin_shufflevector(t0, t1,
                0, 1, 2, 3, 4, 5, 6, 7, 8, 9, 10, 11, 12, 13, 14, 15);
            a0c2 = __builtin_amdgcn_wmma_f32_16x16x32_f16(false, af0, false, bf,
                                                          (short)0, a0c2, false, false);
            a1c2 = __builtin_amdgcn_wmma_f32_16x16x32_f16(false, af1, false, bf,
                                                          (short)0, a1c2, false, false);

            t0 = *(const v8h*)(w0 + (size_t)48 * K);
            t1 = *(const v8h*)(w0 + (size_t)48 * K + 8);
            bf = __builtin_shufflevector(t0, t1,
                0, 1, 2, 3, 4, 5, 6, 7, 8, 9, 10, 11, 12, 13, 14, 15);
            a0c3 = __builtin_amdgcn_wmma_f32_16x16x32_f16(false, af0, false, bf,
                                                          (short)0, a0c3, false, false);
            a1c3 = __builtin_amdgcn_wmma_f32_16x16x32_f16(false, af1, false, bf,
                                                          (short)0, a1c3, false, false);
        }
    }

    // D layout: VGPR r, lanes 0-15 -> M=r, lanes 16-31 -> M=8+r; N = lane&15.
    int cb = tn * 64 + lhalf;
    #pragma unroll
    for (int r = 0; r < 8; ++r) {
        int ms0 = tm * 32 + r + hi * 8;
        int bs0 = ms0 / Vlvl;
        int vs0 = ms0 - bs0 * Vlvl;
        float* orow0 = out + ((size_t)bs0 * Vlvl + vs0) * N + cb;
        orow0[0]  = fmaxf(a0c0[r], 0.0f);
        orow0[16] = fmaxf(a0c1[r], 0.0f);
        orow0[32] = fmaxf(a0c2[r], 0.0f);
        orow0[48] = fmaxf(a0c3[r], 0.0f);

        int ms1 = ms0 + 16;
        int bs1 = ms1 / Vlvl;
        int vs1 = ms1 - bs1 * Vlvl;
        float* orow1 = out + ((size_t)bs1 * Vlvl + vs1) * N + cb;
        orow1[0]  = fmaxf(a1c0[r], 0.0f);
        orow1[16] = fmaxf(a1c1[r], 0.0f);
        orow1[32] = fmaxf(a1c2[r], 0.0f);
        orow1[48] = fmaxf(a1c3[r], 0.0f);
    }
}

// ---------------------------------------------------------------------------
// Head: out[b][v][o] = bh[o] + sum_{l,c} Wh[o][l*64+c] * hin[b][sp0[v][l]][c]
// (B*778*3 outputs, K=576 f32 -- negligible work, VALU kernel)
// ---------------------------------------------------------------------------
__global__ void head_kernel(const float* __restrict__ hin, const int* __restrict__ sp,
                            const float* __restrict__ Wh, const float* __restrict__ bh,
                            float* __restrict__ out, int total) {
    int t = blockIdx.x * blockDim.x + threadIdx.x;
    if (t >= total) return;
    int o  = t % 3;
    int vb = t / 3;
    int v  = vb % 778;
    int b  = vb / 778;
    const float* w = Wh + o * 576;
    float s = bh[o];
    for (int l = 0; l < 9; ++l) {
        const float* r  = hin + ((size_t)b * 778 + sp[v * 9 + l]) * 64;
        const float* wl = w + l * 64;
        #pragma unroll 8
        for (int c = 0; c < 64; ++c) s += wl[c] * r[c];
    }
    out[t] = s;
}

// ---------------------------------------------------------------------------
// Host orchestration
// ---------------------------------------------------------------------------
extern "C" void kernel_launch(void* const* d_in, const int* in_sizes, int n_in,
                              void* d_out, int out_size, void* d_ws, size_t ws_size,
                              hipStream_t stream) {
    (void)in_sizes; (void)n_in; (void)out_size; (void)ws_size;

    const float* uv  = (const float*)d_in[0];
    const float* x   = (const float*)d_in[1];
    const float* up  = (const float*)d_in[2];
    const int* sp0 = (const int*)d_in[3];
    const int* sp1 = (const int*)d_in[4];
    const int* sp2 = (const int*)d_in[5];
    const int* sp3 = (const int*)d_in[6];
    const int*   ui0 = (const int*)d_in[7];   const float* uvl0 = (const float*)d_in[8];
    const int*   ui1 = (const int*)d_in[9];   const float* uvl1 = (const float*)d_in[10];
    const int*   ui2 = (const int*)d_in[11];  const float* uvl2 = (const float*)d_in[12];
    const int*   ui3 = (const int*)d_in[13];  const float* uvl3 = (const float*)d_in[14];
    const float* W0 = (const float*)d_in[15]; const float* b0 = (const float*)d_in[16];
    const float* W1 = (const float*)d_in[17]; const float* b1 = (const float*)d_in[18];
    const float* W2 = (const float*)d_in[19]; const float* b2 = (const float*)d_in[20];
    const float* W3 = (const float*)d_in[21]; const float* b3 = (const float*)d_in[22];
    const float* Wh = (const float*)d_in[23]; const float* bh = (const float*)d_in[24];
    float* outp = (float*)d_out;

    // Workspace layout (bytes)
    const size_t nW0 = 512ull * 4608, nW1 = 256ull * 4608, nW2 = 128ull * 2304, nW3 = 64ull * 1152;
    char* ws = (char*)d_ws;
    _Float16* w16_0 = (_Float16*)ws;
    _Float16* w16_1 = w16_0 + nW0;
    _Float16* w16_2 = w16_1 + nW1;
    _Float16* w16_3 = w16_2 + nW2;
    size_t off = 2 * (nW0 + nW1 + nW2 + nW3);     // bytes, multiple of 256
    float* h0 = (float*)(ws + off);               off += 64ull * 21 * 512 * 4;
    float* h1 = (float*)(ws + off);               off += 64ull * 49 * 512 * 4;
    _Float16* A16 = (_Float16*)(ws + off);        off += 64ull * 195 * 512 * 2;  // max pooled f16
    float* Cout = (float*)(ws + off);             // max 64*98*512 f32

    // 1) convert weights to f16
    cvt_w_kernel<<<(int)((nW0 + 255) / 256), 256, 0, stream>>>(W0, w16_0, (int)nW0);
    cvt_w_kernel<<<(int)((nW1 + 255) / 256), 256, 0, stream>>>(W1, w16_1, (int)nW1);
    cvt_w_kernel<<<(int)((nW2 + 255) / 256), 256, 0, stream>>>(W2, w16_2, (int)nW2);
    cvt_w_kernel<<<(int)((nW3 + 255) / 256), 256, 0, stream>>>(W3, w16_3, (int)nW3);

    // 2) grid-sample + upsample
    sample_kernel<<<dim3(21, 64), 256, 0, stream>>>(uv, x, h0);
    upsample_kernel<<<dim3(49, 64), 128, 0, stream>>>(up, h0, h1);

    // 3) four pool + WMMA-GEMM levels (32x64 tiles per wave, 8 waves/block)
    // j=0 : 49 -> 98, C=512, N=512
    pool_kernel<<<dim3(98, 64), 128, 0, stream>>>(h1, ui3, uvl3, A16, 49, 98, 512);
    {
        int mt = (64 * 98) / 32, nt = 512 / 64, tiles = mt * nt;
        gemm_spiral_kernel<<<(tiles + 7) / 8, 256, 0, stream>>>(A16, sp3, w16_0, b0, Cout,
                                                                98, 512, 512, mt, nt);
    }
    // j=1 : 98 -> 195, C=512, N=256
    pool_kernel<<<dim3(195, 64), 128, 0, stream>>>(Cout, ui2, uvl2, A16, 98, 195, 512);
    {
        int mt = (64 * 195) / 32, nt = 256 / 64, tiles = mt * nt;
        gemm_spiral_kernel<<<(tiles + 7) / 8, 256, 0, stream>>>(A16, sp2, w16_1, b1, Cout,
                                                                195, 512, 256, mt, nt);
    }
    // j=2 : 195 -> 389, C=256, N=128
    pool_kernel<<<dim3(389, 64), 128, 0, stream>>>(Cout, ui1, uvl1, A16, 195, 389, 256);
    {
        int mt = (64 * 389) / 32, nt = 128 / 64, tiles = mt * nt;
        gemm_spiral_kernel<<<(tiles + 7) / 8, 256, 0, stream>>>(A16, sp1, w16_2, b2, Cout,
                                                                389, 256, 128, mt, nt);
    }
    // j=3 : 389 -> 778, C=128, N=64
    pool_kernel<<<dim3(778, 64), 128, 0, stream>>>(Cout, ui0, uvl0, A16, 389, 778, 128);
    {
        int mt = (64 * 778) / 32, nt = 64 / 64, tiles = mt * nt;
        gemm_spiral_kernel<<<(tiles + 7) / 8, 256, 0, stream>>>(A16, sp0, w16_3, b3, Cout,
                                                                778, 128, 64, mt, nt);
    }

    // 4) head
    {
        int total = 64 * 778 * 3;
        head_kernel<<<(total + 255) / 256, 256, 0, stream>>>(Cout, sp0, Wh, bh, outp, total);
    }
}
